// ResetRNN_17016660426957
// MI455X (gfx1250) — matrix operand order repaired and checked by reading, not compiled, not run on hardware
//
#include <hip/hip_runtime.h>

// ResetRNN (GRU scan with episodic resets) for MI455X / gfx1250, wave32.
//
// Phase 1 (full chip): gi = seq @ Wi as a bf16 WMMA GEMM, written to d_ws
//   (needs T*B*3H*4 = 384 MiB of workspace; memory-bound, ~23 us at 23.3 TB/s).
// Phase 2 (16 WGPs): sequential GRU recurrence. Each block owns a 16-batch
//   tile; h lives in registers (C-layout), Wh lives in registers as bf16
//   B-fragments, h is republished per step via a double-buffered padded LDS
//   tile and re-read as WMMA A-fragments (bank-conflict-free ds_load_b128).

#define TT 1024
#define BB 256
#define DD 128
#define HH 128
#define N3 384
#define PAD 136  // halves per LDS row: 128 + 8 pad -> 68-dword stride, no bank conflicts

typedef __attribute__((ext_vector_type(16))) __bf16 v16bf;
typedef __attribute__((ext_vector_type(8)))  float  v8f;

union Frag {
  v16bf        v;
  uint4        q[2];
  unsigned int u[8];
};

__device__ __forceinline__ unsigned int f2bf1(float f) {
  union { float f; unsigned int u; } c; c.f = f;
  return (c.u + 0x7FFFu + ((c.u >> 16) & 1u)) >> 16;  // RNE f32 -> bf16
}
__device__ __forceinline__ unsigned int pk2(float lo, float hi) {
  return f2bf1(lo) | (f2bf1(hi) << 16);
}
__device__ __forceinline__ float sigm(float x) {
  return 1.0f / (1.0f + __expf(-x));
}
__device__ __forceinline__ float tanh_fast(float x) {
  return 1.0f - 2.0f / (1.0f + __expf(2.0f * x));
}
__device__ __forceinline__ v8f wmma_bf16(const Frag& a, const Frag& b, v8f c) {
  return __builtin_amdgcn_wmma_f32_16x16x32_bf16(
      false, a.v, false, b.v, (short)0, c, false, false);
}

// ---------------------------------------------------------------------------
// Phase 1: gi[T*B, 384] = seq[T*B, 128] @ Wi[128, 384]   (bf16 WMMA, f32 acc)
// Grid: (T*B)/16 blocks of 256 threads (8 waves). Block stages its 16x128
// x-tile into LDS as bf16; wave w computes N-tiles {w, w+8, w+16}.
// ---------------------------------------------------------------------------
__global__ __launch_bounds__(256) void gi_gemm_kernel(
    const float* __restrict__ seq, const float* __restrict__ Wi,
    float* __restrict__ gi) {
  __shared__ __align__(16) unsigned short xb[16 * PAD];
  const int tid  = threadIdx.x;
  const int lane = tid & 31;
  const int w    = tid >> 5;
  const int hi   = lane >> 4;
  const int ln   = lane & 15;
  const long mbase = (long)blockIdx.x * 16;

  // Cooperative stage: 16 rows x 128 cols f32 -> bf16 LDS (one b128 store/thread)
  {
    const int r  = tid & 15;
    const int c0 = (tid >> 4) * 8;
    const float* s = seq + (mbase + r) * DD + c0;
    float4 f0 = *(const float4*)(s);
    float4 f1 = *(const float4*)(s + 4);
    uint4 o;
    o.x = pk2(f0.x, f0.y); o.y = pk2(f0.z, f0.w);
    o.z = pk2(f1.x, f1.y); o.w = pk2(f1.z, f1.w);
    *(uint4*)&xb[r * PAD + c0] = o;
  }
  __syncthreads();

  // A fragments (16x32 bf16 each), shared across this wave's 3 N-tiles.
  Frag a[4];
#pragma unroll
  for (int kf = 0; kf < 4; ++kf) {
    const int kb = kf * 32 + 8 * hi;
    a[kf].q[0] = *(const uint4*)&xb[ln * PAD + kb];
    a[kf].q[1] = *(const uint4*)&xb[ln * PAD + kb + 16];
  }

#pragma unroll
  for (int i = 0; i < 3; ++i) {
    const int ncol = (w + 8 * i) * 16;
    Frag b[4];
#pragma unroll
    for (int kf = 0; kf < 4; ++kf)
#pragma unroll
      for (int j = 0; j < 8; ++j) {
        const int k = kf * 32 + 8 * hi + 2 * (j & 3) + (j >> 2) * 16;  // B: table order, matches A
        b[kf].u[j] = pk2(Wi[(long)k * N3 + ncol + ln],
                         Wi[(long)(k + 1) * N3 + ncol + ln]);
      }
    v8f acc = {};
#pragma unroll
    for (int kf = 0; kf < 4; ++kf) acc = wmma_bf16(a[kf], b[kf], acc);
#pragma unroll
    for (int v = 0; v < 8; ++v)  // C layout: VGPR v -> row v+8*hi, lane%16 -> col
      gi[(mbase + v + 8 * hi) * N3 + ncol + ln] = acc[v];
  }
}

// ---------------------------------------------------------------------------
// Phase 2: sequential GRU scan. Grid: B/16 = 16 blocks of 256 threads.
// Wave w owns H-columns [16w, 16w+16); its 3 gate accumulators use N-tiles at
// {16w, 128+16w, 256+16w}. h kept in registers; republished via LDS per step.
// ---------------------------------------------------------------------------
__global__ __launch_bounds__(256) void gru_scan_kernel(
    const float* __restrict__ gi, const unsigned char* __restrict__ resets,
    const float* __restrict__ h0, const float* __restrict__ Wh,
    const float* __restrict__ bh, float* __restrict__ out) {
  __shared__ __align__(16) unsigned short hbuf[2][16 * PAD];
  const int tid  = threadIdx.x;
  const int lane = tid & 31;
  const int w    = tid >> 5;
  const int hi   = lane >> 4;
  const int ln   = lane & 15;
  const int col  = w * 16 + ln;          // owned H column
  const int bbase = blockIdx.x * 16;     // owned batch tile

  // Register-resident Wh B-fragments for the 3 gates (96 VGPRs).
  Frag wb[3][4];
#pragma unroll
  for (int g = 0; g < 3; ++g) {
    const int ncol = g * HH + col;
#pragma unroll
    for (int kf = 0; kf < 4; ++kf)
#pragma unroll
      for (int j = 0; j < 8; ++j) {
        const int k = kf * 32 + 8 * hi + 2 * (j & 3) + (j >> 2) * 16;
        wb[g][kf].u[j] = pk2(Wh[(long)k * N3 + ncol],
                             Wh[(long)(k + 1) * N3 + ncol]);
      }
  }
  const float bhr = bh[col], bhz = bh[HH + col], bhn = bh[2 * HH + col];

  float hc[8], hini[8];
#pragma unroll
  for (int v = 0; v < 8; ++v) {
    hini[v] = h0[(long)(bbase + v + 8 * hi) * HH + col];
    hc[v]   = hini[v];
  }
  float* __restrict__ ys = out + BB * HH;

#pragma unroll 2
  for (int t = 0; t < TT; ++t) {
    // gi pre-activations (off the serial dependency chain: overlaps WMMAs)
    const float* gb = gi + ((long)t * BB + bbase) * N3;
    float gr[8], gz[8], gn[8];
#pragma unroll
    for (int v = 0; v < 8; ++v) {
      const float* row = gb + (long)(v + 8 * hi) * N3 + col;
      gr[v] = row[0]; gz[v] = row[HH]; gn[v] = row[2 * HH];
    }

    // episodic resets: 8 bytes for this lane's 8 batch rows in one b64 load
    unsigned long long rb =
        *(const unsigned long long*)(resets + (long)t * BB + bbase + 8 * hi);
#pragma unroll
    for (int v = 0; v < 8; ++v)
      if ((rb >> (8 * v)) & 1ull) hc[v] = hini[v];

    // publish post-reset h as bf16 (double-buffered -> one barrier per step)
    unsigned short* hb = hbuf[t & 1];
#pragma unroll
    for (int v = 0; v < 8; ++v)
      hb[(v + 8 * hi) * PAD + col] = (unsigned short)f2bf1(hc[v]);
    __syncthreads();

    // A fragments of the h tile (conflict-free ds_load_b128 pairs)
    Frag a[4];
#pragma unroll
    for (int kf = 0; kf < 4; ++kf) {
      const int kb = kf * 32 + 8 * hi;
      a[kf].q[0] = *(const uint4*)&hb[ln * PAD + kb];
      a[kf].q[1] = *(const uint4*)&hb[ln * PAD + kb + 16];
    }

    v8f ar = {}, az = {}, an = {};
#pragma unroll
    for (int kf = 0; kf < 4; ++kf) {
      ar = wmma_bf16(a[kf], wb[0][kf], ar);
      az = wmma_bf16(a[kf], wb[1][kf], az);
      an = wmma_bf16(a[kf], wb[2][kf], an);
    }

    // gates + state update + ys store (C layout: VGPR v -> batch row v+8*hi)
#pragma unroll
    for (int v = 0; v < 8; ++v) {
      float r  = sigm(gr[v] + ar[v] + bhr);
      float z  = sigm(gz[v] + az[v] + bhz);
      float n  = tanh_fast(gn[v] + r * (an[v] + bhn));
      float hn = (1.0f - z) * n + z * hc[v];
      hc[v] = hn;
      ys[((long)t * BB + bbase + v + 8 * hi) * HH + col] = hn;
    }
  }

  // final_carry at the front of d_out
#pragma unroll
  for (int v = 0; v < 8; ++v)
    out[(long)(bbase + v + 8 * hi) * HH + col] = hc[v];
}

extern "C" void kernel_launch(void* const* d_in, const int* in_sizes, int n_in,
                              void* d_out, int out_size, void* d_ws, size_t ws_size,
                              hipStream_t stream) {
  const float*         seq    = (const float*)d_in[0];
  const unsigned char* resets = (const unsigned char*)d_in[1];  // jax bool: 1 byte
  const float*         h0     = (const float*)d_in[2];
  const float*         Wi     = (const float*)d_in[3];
  const float*         Wh     = (const float*)d_in[4];
  const float*         bh     = (const float*)d_in[5];
  float*               out    = (float*)d_out;
  float*               gi     = (float*)d_ws;  // needs T*B*3H*4 = 384 MiB scratch

  (void)in_sizes; (void)n_in; (void)out_size; (void)ws_size;

  gi_gemm_kernel<<<dim3((TT * BB) / 16), dim3(256), 0, stream>>>(seq, Wi, gi);
  gru_scan_kernel<<<dim3(BB / 16), dim3(256), 0, stream>>>(gi, resets, h0, Wh, bh, out);
}
